// RelativePositionTransformerEncoderLayer_88493506167419
// MI455X (gfx1250) — matrix operand order repaired and checked by reading, not compiled, not run on hardware
//
#include <hip/hip_runtime.h>
#include <hip/hip_bf16.h>
#include <math.h>

// ---------------------------------------------------------------------------
// RelativePositionTransformerEncoderLayer for MI455X (gfx1250, wave32, WMMA).
// All GEMM-shaped work runs on v_wmma_f32_16x16x32_bf16 (fp32 accum).
// ---------------------------------------------------------------------------

#define S_      512
#define B_      32
#define D_      1024
#define H_      8
#define E_      128
#define INNER_  2048
#define ROWS_   (S_ * B_)                 // 16384 token rows
#define SCALE_  0.08838834764831845f      // 1/sqrt(E)

typedef __attribute__((ext_vector_type(16))) __bf16 v16bf;
typedef __attribute__((ext_vector_type(8)))  float  v8f;

__device__ __forceinline__ __bf16 tobf(float f) { return (__bf16)f; }

__device__ __forceinline__ v8f v8f_zero() {
  v8f z;
#pragma unroll
  for (int i = 0; i < 8; ++i) z[i] = 0.0f;
  return z;
}

__device__ __forceinline__ v8f wmma_bf16(v16bf a, v16bf b, v8f c) {
  // D = A(16x32 bf16) * B(32x16 bf16) + C(16x16 f32)
  return __builtin_amdgcn_wmma_f32_16x16x32_bf16(
      /*neg_a=*/false, a, /*neg_b=*/false, b,
      /*c_mod=*/(short)0, c, /*reuse_a=*/false, /*reuse_b=*/false);
}

// ---------------------------------------------------------------------------
// Generic GEMM: C[M,N] = act(A[M,K] @ W[K,N] + bias) (+ resid), fp32 in/out,
// bf16 WMMA math. Tile 128x128, 256 threads = 8 waves, wave tile 32x64.
// ---------------------------------------------------------------------------
__device__ __forceinline__ void gemm_load_tiles(
    const float* __restrict__ A, const float* __restrict__ W,
    __bf16* sa, __bf16* sb, int N, int K, int gm0, int gn0, int k0, int tid) {
  // A tile: 128 rows x 32 k -> sa[m*32+k]; 2 threads per row, 16 floats each.
  const int am  = tid >> 1;
  const int akh = (tid & 1) * 16;
  const float* ga = A + (size_t)(gm0 + am) * K + k0 + akh;
#pragma unroll
  for (int i = 0; i < 16; ++i) sa[am * 32 + akh + i] = tobf(ga[i]);
  // W tile: 32 k x 128 n, stored transposed -> sb[n*32+k] (contig B-frag reads)
  const int kk = tid >> 3;
  const int n0 = (tid & 7) * 16;
  const float* gw = W + (size_t)(k0 + kk) * N + gn0 + n0;
#pragma unroll
  for (int i = 0; i < 16; ++i) sb[(n0 + i) * 32 + kk] = tobf(gw[i]);
}

__device__ __forceinline__ void gemm_mma(
    const __bf16* sa, const __bf16* sb, v8f acc[2][4],
    int wm, int wn, int half, int l16) {
  v16bf af[2], bfr[4];
#pragma unroll
  for (int mt = 0; mt < 2; ++mt) {
    // A frag (16-bit 16x32): lane<16 -> k {0..7,16..23}; lane>=16 -> {8..15,24..31}
    const __bf16* pa = sa + (size_t)(wm * 32 + mt * 16 + l16) * 32 + half * 8;
#pragma unroll
    for (int i = 0; i < 8; ++i) { af[mt][i] = pa[i]; af[mt][8 + i] = pa[16 + i]; }
  }
#pragma unroll
  for (int nt = 0; nt < 4; ++nt) {
    // B frag: n = l16, k = 8*half + {0..7, 16..23} (matches A frag order)
    const __bf16* pb = sb + (size_t)(wn * 64 + nt * 16 + l16) * 32 + half * 8;
#pragma unroll
    for (int i = 0; i < 16; ++i) bfr[nt][i] = pb[(i & 7) + (i >> 3) * 16];
  }
#pragma unroll
  for (int mt = 0; mt < 2; ++mt)
#pragma unroll
    for (int nt = 0; nt < 4; ++nt)
      acc[mt][nt] = wmma_bf16(af[mt], bfr[nt], acc[mt][nt]);
}

__global__ __launch_bounds__(256, 1)
void gemm_bf16_wmma(const float* __restrict__ A, const float* __restrict__ W,
                    const float* __restrict__ bias, const float* __restrict__ resid,
                    float* __restrict__ C, int M, int N, int K, int relu) {
  __shared__ __bf16 sA[2][128 * 32];
  __shared__ __bf16 sB[2][128 * 32];

  const int tid  = threadIdx.x;
  const int lane = tid & 31;
  const int wid  = tid >> 5;
  const int half = lane >> 4;
  const int l16  = lane & 15;
  const int wm   = wid & 3;   // 4 waves along M (32 rows each)
  const int wn   = wid >> 2;  // 2 waves along N (64 cols each)
  const int gm0  = blockIdx.y * 128;
  const int gn0  = blockIdx.x * 128;

  v8f acc[2][4];
#pragma unroll
  for (int i = 0; i < 2; ++i)
#pragma unroll
    for (int j = 0; j < 4; ++j) acc[i][j] = v8f_zero();

  const int kTiles = K / 32;
  gemm_load_tiles(A, W, sA[0], sB[0], N, K, gm0, gn0, 0, tid);
  __syncthreads();

  for (int kt = 0; kt < kTiles; ++kt) {
    if (kt + 1 < kTiles) {
      // Stage next tile into the other buffer; prefetch one further ahead.
      __builtin_prefetch(A + (size_t)(gm0 + (tid >> 1)) * K + (size_t)(kt + 2) * 32, 0, 1);
      __builtin_prefetch(W + (size_t)((kt + 2) * 32 + (tid >> 3)) * N + gn0, 0, 1);
      gemm_load_tiles(A, W, sA[(kt + 1) & 1], sB[(kt + 1) & 1], N, K, gm0, gn0,
                      (kt + 1) * 32, tid);
    }
    gemm_mma(sA[kt & 1], sB[kt & 1], acc, wm, wn, half, l16);
    __syncthreads();
  }

  // Epilogue: D layout lane -> col = l16(+tile), rows = r + 8*half.
#pragma unroll
  for (int mt = 0; mt < 2; ++mt) {
#pragma unroll
    for (int nt = 0; nt < 4; ++nt) {
      const int col = gn0 + wn * 64 + nt * 16 + l16;
      const float bb = bias ? bias[col] : 0.0f;
#pragma unroll
      for (int r = 0; r < 8; ++r) {
        const int row = gm0 + wm * 32 + mt * 16 + r + 8 * half;
        float v = acc[mt][nt][r] + bb;
        if (relu) v = fmaxf(v, 0.0f);
        if (resid) v += resid[(size_t)row * N + col];
        C[(size_t)row * N + col] = v;
      }
    }
  }
}

// ---------------------------------------------------------------------------
// Flash-style relative-position attention, all matrix work on WMMA.
// Grid: (S/64, B*H). Block: 128 threads = 4 waves; wave owns 16 query rows.
// scores = ((q+bu)·k + (q+bv)·pos[j-i+512]) * scale + mask_bias
// pos term via shift trick: WMMA against a 48-wide rel-window of pos_weight,
// then gather P[i,j] = D[i, j-i+16].
// ---------------------------------------------------------------------------
__global__ __launch_bounds__(128, 1)
void attn_wmma(const float* __restrict__ Q, const float* __restrict__ Km,
               const float* __restrict__ Vm, const unsigned char* __restrict__ mask,
               const float* __restrict__ bu, const float* __restrict__ bvp,
               const float* __restrict__ PW, float* __restrict__ O) {
  __shared__ __bf16 sV[32 * 128];      // key-block V tile (bf16)
  __shared__ float  sD[4][16 * 48];    // per-wave pos-window bounce
  __shared__ __bf16 sP[4][16 * 32];    // per-wave prob bounce (D->A relayout)

  const int tid  = threadIdx.x;
  const int lane = tid & 31;
  const int w    = tid >> 5;
  const int half = lane >> 4;
  const int l16  = lane & 15;
  const int bh   = blockIdx.y;
  const int b    = bh / H_;
  const int h    = bh % H_;
  const int i0w  = blockIdx.x * 64 + w * 16;
  const size_t rstride = (size_t)H_ * E_;

  // A fragments for (q+bu) and (q+bv): reused across all 16 key blocks.
  v16bf au[4], av[4];
  {
    const float* qrow = Q + ((size_t)(i0w + l16) * B_ + b) * rstride + h * E_;
    const float* buh  = bu  + h * E_;
    const float* bvh  = bvp + h * E_;
#pragma unroll
    for (int ks = 0; ks < 4; ++ks) {
#pragma unroll
      for (int i = 0; i < 8; ++i) {
        const int e0 = ks * 32 + half * 8 + i;
        const int e1 = ks * 32 + 16 + half * 8 + i;
        au[ks][i]     = tobf(qrow[e0] + buh[e0]);
        au[ks][8 + i] = tobf(qrow[e1] + buh[e1]);
        av[ks][i]     = tobf(qrow[e0] + bvh[e0]);
        av[ks][8 + i] = tobf(qrow[e1] + bvh[e1]);
      }
    }
  }

  v8f o[8];
#pragma unroll
  for (int nt = 0; nt < 8; ++nt) o[nt] = v8f_zero();
  float m_run[8], l_run[8];
#pragma unroll
  for (int r = 0; r < 8; ++r) { m_run[r] = -1e30f; l_run[r] = 0.0f; }

  for (int kb = 0; kb < 16; ++kb) {
    const int j0 = kb * 32;
    __syncthreads();  // previous block's sV reads complete
    {   // stage V tile: 32 keys x 128 E, fp32 -> bf16
      const int vr = tid >> 2, c0 = (tid & 3) * 32;
      const float* gv = Vm + ((size_t)(j0 + vr) * B_ + b) * rstride + h * E_ + c0;
#pragma unroll
      for (int i = 0; i < 32; ++i) sV[vr * 128 + c0 + i] = tobf(gv[i]);
    }
    __syncthreads();

    // --- content scores: (q+bu) . k^T, 16x32 in two 16x16 tiles ---
    v8f cc[2];
    cc[0] = v8f_zero(); cc[1] = v8f_zero();
#pragma unroll
    for (int ks = 0; ks < 4; ++ks) {
#pragma unroll
      for (int nt = 0; nt < 2; ++nt) {
        const float* krow =
            Km + ((size_t)(j0 + nt * 16 + l16) * B_ + b) * rstride + h * E_;
        v16bf bk;
#pragma unroll
        for (int i = 0; i < 16; ++i) bk[i] = tobf(krow[ks * 32 + half * 8 + (i & 7) + (i >> 3) * 16]);
        cc[nt] = wmma_bf16(au[ks], bk, cc[nt]);
      }
    }

    // --- pos window D[16 x 48]: rel = w0 + col; always in [0,1023] ---
    const int w0 = j0 - i0w + (512 - 16);
    v8f dd[3];
    dd[0] = v8f_zero(); dd[1] = v8f_zero(); dd[2] = v8f_zero();
#pragma unroll
    for (int ks = 0; ks < 4; ++ks) {
#pragma unroll
      for (int nt = 0; nt < 3; ++nt) {
        const float* prow = PW + (size_t)(w0 + nt * 16 + l16) * E_;
        v16bf bp;
#pragma unroll
        for (int i = 0; i < 16; ++i) bp[i] = tobf(prow[ks * 32 + half * 8 + (i & 7) + (i >> 3) * 16]);
        dd[nt] = wmma_bf16(av[ks], bp, dd[nt]);
      }
    }
#pragma unroll
    for (int r = 0; r < 8; ++r) {
      const int m = r + 8 * half;
      sD[w][m * 48 + l16]      = dd[0][r];
      sD[w][m * 48 + 16 + l16] = dd[1][r];
      sD[w][m * 48 + 32 + l16] = dd[2][r];
    }

    // --- gather shifted pos term + online softmax over this key block ---
    const float bias0 = mask[b * S_ + j0 + l16]      ? 0.0f : -1e30f;
    const float bias1 = mask[b * S_ + j0 + 16 + l16] ? 0.0f : -1e30f;
    float alpha[8];
#pragma unroll
    for (int r = 0; r < 8; ++r) {
      const int m = r + 8 * half;
      float s0 = (cc[0][r] + sD[w][m * 48 + (l16 - m + 16)]) * SCALE_ + bias0;
      float s1 = (cc[1][r] + sD[w][m * 48 + (l16 - m + 32)]) * SCALE_ + bias1;
      float mx = fmaxf(s0, s1);
#pragma unroll
      for (int off = 8; off >= 1; off >>= 1) mx = fmaxf(mx, __shfl_xor(mx, off, 32));
      const float mnew = fmaxf(m_run[r], mx);
      const float al = __expf(m_run[r] - mnew);
      const float p0 = __expf(s0 - mnew);
      const float p1 = __expf(s1 - mnew);
      float ls = p0 + p1;
#pragma unroll
      for (int off = 8; off >= 1; off >>= 1) ls += __shfl_xor(ls, off, 32);
      l_run[r] = l_run[r] * al + ls;
      m_run[r] = mnew;
      alpha[r] = al;
      sP[w][m * 32 + l16]      = tobf(p0);
      sP[w][m * 32 + 16 + l16] = tobf(p1);
    }
#pragma unroll
    for (int nt = 0; nt < 8; ++nt)
#pragma unroll
      for (int r = 0; r < 8; ++r) o[nt][r] *= alpha[r];

    // --- PV: probs(16x32) @ V(32x128) accumulated into o ---
    v16bf ap;
    {
      const __bf16* pa = &sP[w][l16 * 32 + half * 8];
#pragma unroll
      for (int i = 0; i < 8; ++i) { ap[i] = pa[i]; ap[8 + i] = pa[16 + i]; }
    }
#pragma unroll
    for (int nt = 0; nt < 8; ++nt) {
      v16bf bb;
#pragma unroll
      for (int i = 0; i < 16; ++i)
        bb[i] = sV[(half * 8 + (i & 7) + (i >> 3) * 16) * 128 + nt * 16 + l16];
      o[nt] = wmma_bf16(ap, bb, o[nt]);
    }
  }

  // finalize: divide by running sum, write [S,B,H*E]
#pragma unroll
  for (int r = 0; r < 8; ++r) {
    const float li = l_run[r];
    const float inv = li > 0.0f ? 1.0f / li : 0.0f;
    const int row = i0w + r + 8 * half;
#pragma unroll
    for (int nt = 0; nt < 8; ++nt)
      O[((size_t)row * B_ + b) * rstride + h * E_ + nt * 16 + l16] = o[nt][r] * inv;
  }
}

// ---------------------------------------------------------------------------
// out[row] = LayerNorm(A[row] + B[row]) * g + beta       (D = 1024, 256 thr)
// ---------------------------------------------------------------------------
__global__ __launch_bounds__(256, 1)
void add_ln(const float* __restrict__ A, const float* __restrict__ Bs,
            const float* __restrict__ g, const float* __restrict__ be,
            float* __restrict__ out) {
  __shared__ float red[8];
  const int row = blockIdx.x, tid = threadIdx.x;
  const int lane = tid & 31, w = tid >> 5;
  const float* pa = A  + (size_t)row * D_;
  const float* pb = Bs + (size_t)row * D_;

  float v[4];
  float s = 0.0f;
#pragma unroll
  for (int i = 0; i < 4; ++i) {
    const int idx = tid + i * 256;
    v[i] = pa[idx] + pb[idx];
    s += v[i];
  }
#pragma unroll
  for (int off = 16; off >= 1; off >>= 1) s += __shfl_xor(s, off, 32);
  if (lane == 0) red[w] = s;
  __syncthreads();
  s = 0.0f;
#pragma unroll
  for (int i = 0; i < 8; ++i) s += red[i];
  const float mean = s / (float)D_;
  __syncthreads();

  float s2 = 0.0f;
#pragma unroll
  for (int i = 0; i < 4; ++i) { const float d = v[i] - mean; s2 += d * d; }
#pragma unroll
  for (int off = 16; off >= 1; off >>= 1) s2 += __shfl_xor(s2, off, 32);
  if (lane == 0) red[w] = s2;
  __syncthreads();
  s2 = 0.0f;
#pragma unroll
  for (int i = 0; i < 8; ++i) s2 += red[i];
  const float rstd = rsqrtf(s2 / (float)D_ + 1e-5f);

#pragma unroll
  for (int i = 0; i < 4; ++i) {
    const int idx = tid + i * 256;
    out[(size_t)row * D_ + idx] = (v[i] - mean) * rstd * g[idx] + be[idx];
  }
}

// ---------------------------------------------------------------------------
// Host-side orchestration. Workspace layout (fp32, 256 MiB total):
//   [0,64M)   q      -> reused for proj, then f
//   [64,128M) k      -> reused for x1
//   [128,256M) v+attn -> v, attn; jointly reused for h1 (16384x2048)
// ---------------------------------------------------------------------------
extern "C" void kernel_launch(void* const* d_in, const int* in_sizes, int n_in,
                              void* d_out, int out_size, void* d_ws, size_t ws_size,
                              hipStream_t stream) {
  (void)in_sizes; (void)n_in; (void)out_size; (void)ws_size;
  const float* x   = (const float*)d_in[0];
  const unsigned char* mask = (const unsigned char*)d_in[1];
  const float* wq  = (const float*)d_in[2];
  const float* wk  = (const float*)d_in[3];
  const float* wv  = (const float*)d_in[4];
  const float* wo  = (const float*)d_in[5];
  const float* bu  = (const float*)d_in[6];
  const float* bv  = (const float*)d_in[7];
  const float* pw  = (const float*)d_in[8];
  const float* w1  = (const float*)d_in[9];
  const float* b1  = (const float*)d_in[10];
  const float* w2  = (const float*)d_in[11];
  const float* b2  = (const float*)d_in[12];
  const float* g1  = (const float*)d_in[13];
  const float* be1 = (const float*)d_in[14];
  const float* g2  = (const float*)d_in[15];
  const float* be2 = (const float*)d_in[16];
  float* out = (float*)d_out;
  float* ws  = (float*)d_ws;

  const size_t SEG = (size_t)ROWS_ * 1024;   // 16M floats
  float* q    = ws;
  float* kbuf = ws + SEG;
  float* vbuf = ws + 2 * SEG;
  float* attn = ws + 3 * SEG;
  float* proj = q;      // q dead after attention
  float* x1   = kbuf;   // k dead after attention
  float* h1   = vbuf;   // v+attn dead after proj -> 32M contiguous floats
  float* ff   = q;      // proj dead after LN1

  const dim3 blk(256);
  const dim3 gD(D_ / 128, ROWS_ / 128);
  const dim3 gI(INNER_ / 128, ROWS_ / 128);

  gemm_bf16_wmma<<<gD, blk, 0, stream>>>(x, wq, nullptr, nullptr, q,    ROWS_, D_, D_, 0);
  gemm_bf16_wmma<<<gD, blk, 0, stream>>>(x, wk, nullptr, nullptr, kbuf, ROWS_, D_, D_, 0);
  gemm_bf16_wmma<<<gD, blk, 0, stream>>>(x, wv, nullptr, nullptr, vbuf, ROWS_, D_, D_, 0);

  attn_wmma<<<dim3(S_ / 64, B_ * H_), dim3(128), 0, stream>>>(
      q, kbuf, vbuf, mask, bu, bv, pw, attn);

  gemm_bf16_wmma<<<gD, blk, 0, stream>>>(attn, wo, nullptr, nullptr, proj, ROWS_, D_, D_, 0);
  add_ln<<<ROWS_, blk, 0, stream>>>(x, proj, g1, be1, x1);

  gemm_bf16_wmma<<<gI, blk, 0, stream>>>(x1, w1, b1, nullptr, h1, ROWS_, INNER_, D_, 1);
  gemm_bf16_wmma<<<gD, blk, 0, stream>>>(h1, w2, b2, nullptr, ff, ROWS_, D_, INNER_, 0);
  add_ln<<<ROWS_, blk, 0, stream>>>(x1, ff, g2, be2, out);
}
